// AR_TSP_net_13005160973049
// MI455X (gfx1250) — matrix operand order repaired and checked by reading, not compiled, not run on hardware
//
#include <hip/hip_runtime.h>

// ---------------------------------------------------------------------------
// AR TSP pointer-network forward for MI455X (gfx1250, wave32, WMMA).
// GEMMs: v_wmma_f32_16x16x32_bf16 (fp32 -> bf16 inputs, f32 accumulate),
// guard-free exact tiling, double-buffered LDS (one barrier per K-step),
// fragment loads issued before next-tile staging so WMMA waits only on its
// own ds loads, not on next-tile ds stores.
// Attention / softmax / layernorm / pointer head are VALU kernels with
// LDS-tree reductions (wave32-safe).
// ---------------------------------------------------------------------------

#define BSZ   512
#define SEQ   50
#define TOKN  51              // SEQ + start token
#define TOK   (BSZ * TOKN)    // 26112 encoder tokens = 408 * 64
#define DMODEL 128
#define NHEAD 8
#define HD    16
#define FF    512
#define NEGV  (-1.0e9f)

typedef __attribute__((ext_vector_type(16))) __bf16 v16bf;
typedef __attribute__((ext_vector_type(8)))  float  v8f;

static __device__ __forceinline__ __bf16 f2bf(float f) {
  union { float f; unsigned u; } v; v.f = f;
  unsigned r = v.u + 0x7FFFu + ((v.u >> 16) & 1u);   // round-to-nearest-even
  unsigned short h = (unsigned short)(r >> 16);
  return __builtin_bit_cast(__bf16, h);
}
// packed bf16 pair: [15:0] = lo, [31:16] = hi (RNE)
static __device__ __forceinline__ unsigned bfpair(float lo, float hi) {
  union { float f; unsigned u; } a, b;
  a.f = lo; b.f = hi;
  unsigned ra = (a.u + 0x7FFFu + ((a.u >> 16) & 1u)) >> 16;
  unsigned rb = (b.u + 0x7FFFu + ((b.u >> 16) & 1u)) & 0xFFFF0000u;
  return ra | rb;
}

struct B16x2 { uint4 a, b; };
static __device__ __forceinline__ v16bf ld_frag(const __bf16* p0, const __bf16* p1) {
  B16x2 t;
  t.a = *(const uint4*)p0;
  t.b = *(const uint4*)p1;
  return __builtin_bit_cast(v16bf, t);
}

// ---------------------------------------------------------------------------
// Tiled WMMA GEMM:  C[M,N] = act(A[M,K] @ W[K,N] + bias[N])
// REQUIRES: M % 64 == 0, N % 64 == 0, K % 32 == 0 (true for every GEMM here).
// Block tile 64(M) x 64(N); 256 threads = 8 waves in 4x2; each wave owns a
// 16x32 strip -> 2 WMMAs per K-step sharing one A fragment.
// Double-buffered LDS, one barrier per K-step; fragment ds loads issued
// before next-tile staging stores. Row stride 40 halves (80 B): 16-B
// aligned, conflict-free b128 reads.
// ---------------------------------------------------------------------------
__global__ __launch_bounds__(256)
void gemm_bf16_wmma(const float* __restrict__ A, int lda,
                    const float* __restrict__ W, int ldw,
                    const float* __restrict__ bias,
                    float* __restrict__ C, int ldc,
                    int K, int act)
{
  __shared__ __align__(16) __bf16 As[2][64][40];   // [buf][m][k]
  __shared__ __align__(16) __bf16 Bs[2][64][40];   // [buf][n][k] (W transposed)

  const int tid  = threadIdx.x;
  const int lane = tid & 31;
  const int wid  = tid >> 5;        // 0..7
  const int wm   = wid & 3;         // M tile 0..3
  const int wn   = wid >> 2;        // N strip 0..1
  const int half = lane >> 4;       // 0/1
  const int lrow = lane & 15;

  const int m0 = blockIdx.x * 64;
  const int n0 = blockIdx.y * 64;

  // A staging: 64x32 floats = 2 float4 per thread
  const int ar  = tid >> 3;         // 0..31
  const int ac4 = (tid & 7) * 4;    // 0,4,..,28
  // B staging: 32(k) x 64(n) floats = 2 float4 per thread
  const int bk  = tid >> 4;         // 0..15
  const int bn4 = (tid & 15) * 4;   // 0,4,..,60

  const float* Ap0 = A + (long)(m0 + ar) * lda + ac4;
  const float* Ap1 = A + (long)(m0 + ar + 32) * lda + ac4;
  const float* Wp0 = W + (long)bk * ldw + n0 + bn4;
  const float* Wp1 = W + (long)(bk + 16) * ldw + n0 + bn4;

  auto stage = [&](int k0, int pb) {
    const float4 f0 = *(const float4*)(Ap0 + k0);
    const float4 f1 = *(const float4*)(Ap1 + k0);
    *(uint2*)&As[pb][ar][ac4]      = make_uint2(bfpair(f0.x, f0.y), bfpair(f0.z, f0.w));
    *(uint2*)&As[pb][ar + 32][ac4] = make_uint2(bfpair(f1.x, f1.y), bfpair(f1.z, f1.w));
    const float4 g0 = *(const float4*)(Wp0 + (long)k0 * ldw);
    const float4 g1 = *(const float4*)(Wp1 + (long)k0 * ldw);
    Bs[pb][bn4 + 0][bk] = f2bf(g0.x);
    Bs[pb][bn4 + 1][bk] = f2bf(g0.y);
    Bs[pb][bn4 + 2][bk] = f2bf(g0.z);
    Bs[pb][bn4 + 3][bk] = f2bf(g0.w);
    Bs[pb][bn4 + 0][bk + 16] = f2bf(g1.x);
    Bs[pb][bn4 + 1][bk + 16] = f2bf(g1.y);
    Bs[pb][bn4 + 2][bk + 16] = f2bf(g1.z);
    Bs[pb][bn4 + 3][bk + 16] = f2bf(g1.w);
  };

  v8f acc0 = {0.f,0.f,0.f,0.f,0.f,0.f,0.f,0.f};
  v8f acc1 = {0.f,0.f,0.f,0.f,0.f,0.f,0.f,0.f};

  stage(0, 0);
  __syncthreads();

  const int arow = wm * 16 + lrow;
  const int bc   = wn * 32 + lrow;

  int p = 0;
  for (int k0 = 0; ; ) {
    const int kn = k0 + 32;

    // Issue this tile's fragment loads FIRST (buffer p) so the WMMAs only
    // wait on these ds loads, not on the staging stores below.
    // A 16x32 fragment: lanes 0-15 hold K 0..15; lanes 16-31: K 16..31
    v16bf af = ld_frag(&As[p][arow][half * 16], &As[p][arow][half * 16 + 8]);
    // B 32x16 fragments: lane n holds column n; lanes 0-15: K 0..15, 16-31: K 16..31
    v16bf b0 = ld_frag(&Bs[p][bc][half * 16],      &Bs[p][bc][half * 16 + 8]);
    v16bf b1 = ld_frag(&Bs[p][bc + 16][half * 16], &Bs[p][bc + 16][half * 16 + 8]);

    if (kn < K) {
      stage(kn, p ^ 1);                              // fill other buffer
      if (kn + 32 < K)
        __builtin_prefetch(Ap0 + kn + 32, 0, 1);     // global_prefetch_b8
    }

    acc0 = __builtin_amdgcn_wmma_f32_16x16x32_bf16(false, af, false, b0,
                                                   (short)0, acc0, false, false);
    acc1 = __builtin_amdgcn_wmma_f32_16x16x32_bf16(false, af, false, b1,
                                                   (short)0, acc1, false, false);
    if (kn >= K) break;
    __syncthreads();
    p ^= 1;
    k0 = kn;
  }

  // C layout: lanes 0-15 -> M = r, lanes 16-31 -> M = 8+r; N = lane%16
  const int col0  = n0 + wn * 32 + lrow;
  const int rbase = m0 + wm * 16 + half * 8;
  const float bv0 = bias[col0];
  const float bv1 = bias[col0 + 16];
#pragma unroll
  for (int r = 0; r < 8; ++r) {
    float v0 = acc0[r] + bv0;
    float v1 = acc1[r] + bv1;
    if (act == 1) { v0 = fmaxf(v0, 0.f); v1 = fmaxf(v1, 0.f); }
    C[(long)(rbase + r) * ldc + col0]      = v0;
    C[(long)(rbase + r) * ldc + col0 + 16] = v1;
  }
}

// ---------------------------------------------------------------------------
// Embedding + start token
// ---------------------------------------------------------------------------
__global__ void embed_kernel(const float* __restrict__ x,
                             const float* __restrict__ w,   // (2,128)
                             const float* __restrict__ b,
                             const float* __restrict__ start,
                             float* __restrict__ h)
{
  long i = (long)blockIdx.x * blockDim.x + threadIdx.x;
  if (i >= (long)TOK * DMODEL) return;
  int d = (int)(i & 127);
  long tok = i >> 7;
  int n = (int)(tok % TOKN);
  long bb = tok / TOKN;
  if (n < SEQ) {
    const float* xp = x + (bb * SEQ + n) * 2;
    h[i] = xp[0] * w[d] + xp[1] * w[DMODEL + d] + b[d];
  } else {
    h[i] = start[d];
  }
}

// ---------------------------------------------------------------------------
// Encoder attention: one block per (batch, head); two-pass softmax (no spill)
// ---------------------------------------------------------------------------
__global__ __launch_bounds__(64)
void attn_enc_kernel(const float* __restrict__ qkv,   // (TOK, 384)
                     float* __restrict__ out)         // (TOK, 128)
{
  const int bh = blockIdx.x;
  const int b  = bh >> 3, hh = bh & 7;
  __shared__ float q[TOKN][HD], k[TOKN][HD], v[TOKN][HD];
  const int tid = threadIdx.x;
  for (int i = tid; i < TOKN * HD; i += 64) {
    int n = i >> 4, d = i & 15;
    const float* base = qkv + ((long)(b * TOKN + n)) * 384 + hh * HD + d;
    q[n][d] = base[0];
    k[n][d] = base[DMODEL];
    v[n][d] = base[2 * DMODEL];
  }
  __syncthreads();
  if (tid < TOKN) {
    float mx = -1e30f;
#pragma unroll 1
    for (int j = 0; j < TOKN; ++j) {
      float a = 0.f;
#pragma unroll
      for (int d = 0; d < HD; ++d) a += q[tid][d] * k[j][d];
      mx = fmaxf(mx, a * 0.25f);
    }
    float sum = 0.f;
    float o[HD];
#pragma unroll
    for (int d = 0; d < HD; ++d) o[d] = 0.f;
#pragma unroll 1
    for (int j = 0; j < TOKN; ++j) {
      float a = 0.f;
#pragma unroll
      for (int d = 0; d < HD; ++d) a += q[tid][d] * k[j][d];
      float e = __expf(a * 0.25f - mx);
      sum += e;
#pragma unroll
      for (int d = 0; d < HD; ++d) o[d] += e * v[j][d];
    }
    float inv = 1.f / sum;
    float* op = out + ((long)(b * TOKN + tid)) * DMODEL + hh * HD;
#pragma unroll
    for (int d = 0; d < HD; ++d) op[d] = o[d] * inv;
  }
}

// ---------------------------------------------------------------------------
// h = LayerNorm(h + delta) * g + b   (in place, one block per token row)
// ---------------------------------------------------------------------------
__global__ __launch_bounds__(128)
void add_ln_kernel(float* __restrict__ h, const float* __restrict__ delta,
                   const float* __restrict__ g, const float* __restrict__ bta)
{
  const long row = blockIdx.x;
  const int d = threadIdx.x;
  __shared__ float red[128];
  float x = h[row * DMODEL + d] + delta[row * DMODEL + d];
  red[d] = x; __syncthreads();
  for (int s = 64; s > 0; s >>= 1) { if (d < s) red[d] += red[d + s]; __syncthreads(); }
  float mean = red[0] * (1.f / DMODEL); __syncthreads();
  float c = x - mean;
  red[d] = c * c; __syncthreads();
  for (int s = 64; s > 0; s >>= 1) { if (d < s) red[d] += red[d + s]; __syncthreads(); }
  float r = rsqrtf(red[0] * (1.f / DMODEL) + 1e-5f);
  h[row * DMODEL + d] = c * r * g[d] + bta[d];
}

// ---------------------------------------------------------------------------
// Decoder causal self-attention over the KV cache (keys 0..t)
// ---------------------------------------------------------------------------
__global__ __launch_bounds__(64)
void dec_self_attn(const float* __restrict__ q,    // (512,128)
                   const float* __restrict__ Kc,   // (512,50,128)
                   const float* __restrict__ Vc,
                   float* __restrict__ out, int t)
{
  const int bh = blockIdx.x;
  const int b = bh >> 3, hh = bh & 7;
  __shared__ float qv[HD];
  __shared__ float p[64];
  __shared__ float red[64];
  const int tid = threadIdx.x;
  if (tid < HD) qv[tid] = q[(long)b * DMODEL + hh * HD + tid];
  __syncthreads();
  const int nk = t + 1;
  float s = -1e30f;
  if (tid < nk) {
    const float* kr = Kc + ((long)b * SEQ + tid) * DMODEL + hh * HD;
    float a = 0.f;
#pragma unroll
    for (int d = 0; d < HD; ++d) a += qv[d] * kr[d];
    s = a * 0.25f;
  }
  red[tid] = s; __syncthreads();
  for (int st = 32; st > 0; st >>= 1) { if (tid < st) red[tid] = fmaxf(red[tid], red[tid + st]); __syncthreads(); }
  float mx = red[0]; __syncthreads();
  float e = (tid < nk) ? __expf(s - mx) : 0.f;
  p[tid] = e;
  red[tid] = e; __syncthreads();
  for (int st = 32; st > 0; st >>= 1) { if (tid < st) red[tid] += red[tid + st]; __syncthreads(); }
  float inv = 1.f / red[0];
  if (tid < HD) {
    float a = 0.f;
    for (int j = 0; j < nk; ++j)
      a += p[j] * Vc[((long)b * SEQ + j) * DMODEL + hh * HD + tid];
    out[(long)b * DMODEL + hh * HD + tid] = a * inv;
  }
}

// ---------------------------------------------------------------------------
// Decoder cross-attention over encoder memory (51 keys, masked); layer-0
// slice (cols 0..127) of (TOK,256) Katt/Vatt.
// ---------------------------------------------------------------------------
__global__ __launch_bounds__(64)
void dec_cross_attn(const float* __restrict__ q,
                    const float* __restrict__ Katt,
                    const float* __restrict__ Vatt,
                    const int* __restrict__ mask,
                    float* __restrict__ out)
{
  const int bh = blockIdx.x;
  const int b = bh >> 3, hh = bh & 7;
  __shared__ float qv[HD];
  __shared__ float p[64];
  __shared__ float red[64];
  const int tid = threadIdx.x;
  if (tid < HD) qv[tid] = q[(long)b * DMODEL + hh * HD + tid];
  __syncthreads();
  float s = -1e30f;
  if (tid < TOKN) {
    const float* kr = Katt + ((long)b * TOKN + tid) * 256 + hh * HD;
    float a = 0.f;
#pragma unroll
    for (int d = 0; d < HD; ++d) a += qv[d] * kr[d];
    s = mask[b * TOKN + tid] ? NEGV : a * 0.25f;
  }
  red[tid] = s; __syncthreads();
  for (int st = 32; st > 0; st >>= 1) { if (tid < st) red[tid] = fmaxf(red[tid], red[tid + st]); __syncthreads(); }
  float mx = red[0]; __syncthreads();
  float e = (tid < TOKN) ? __expf(s - mx) : 0.f;
  p[tid] = e;
  red[tid] = e; __syncthreads();
  for (int st = 32; st > 0; st >>= 1) { if (tid < st) red[tid] += red[tid + st]; __syncthreads(); }
  float inv = 1.f / red[0];
  if (tid < HD) {
    float a = 0.f;
    for (int j = 0; j < TOKN; ++j)
      a += p[j] * Vatt[((long)b * TOKN + j) * 256 + hh * HD + tid];
    out[(long)b * DMODEL + hh * HD + tid] = a * inv;
  }
}

// ---------------------------------------------------------------------------
// Pointer head: s = mask? NEG : 10*tanh(q.Kf/sqrt(128)); prob=softmax(s/0.1);
// idx = argmax; logp += log prob[idx]; mask[idx]=1; h_t = h_enc[idx] + PE[t+1]
// Kf = cols 128..255 of Katt.
// ---------------------------------------------------------------------------
__global__ __launch_bounds__(64)
void pointer_kernel(const float* __restrict__ q,
                    const float* __restrict__ Katt,
                    int* __restrict__ mask,
                    const float* __restrict__ h_enc,
                    const float* __restrict__ pe,
                    float* __restrict__ h_t,
                    float* __restrict__ d_out, int t)
{
  const int b = blockIdx.x;
  const int tid = threadIdx.x;
  __shared__ float qs[DMODEL];
  __shared__ float sv[64];
  __shared__ int   si[64];
  __shared__ float red[64];
  qs[tid] = q[(long)b * DMODEL + tid];
  qs[tid + 64] = q[(long)b * DMODEL + 64 + tid];
  __syncthreads();
  float s = -1e30f;
  if (tid < TOKN) {
    if (mask[b * TOKN + tid]) {
      s = NEGV;
    } else {
      const float* kr = Katt + ((long)b * TOKN + tid) * 256 + 128;
      float a = 0.f;
      for (int d = 0; d < DMODEL; ++d) a += qs[d] * kr[d];
      s = 10.f * tanhf(a * 0.08838834764831845f);   // 1/sqrt(128)
    }
  }
  sv[tid] = s; si[tid] = tid; __syncthreads();
  for (int st = 32; st > 0; st >>= 1) {
    if (tid < st && sv[tid + st] > sv[tid]) { sv[tid] = sv[tid + st]; si[tid] = si[tid + st]; }
    __syncthreads();
  }
  const int idx = si[0];
  const float mx = sv[0];
  float e = (tid < TOKN) ? __expf((s - mx) * 10.f) : 0.f;   // temperature 0.1
  red[tid] = e; __syncthreads();
  for (int st = 32; st > 0; st >>= 1) { if (tid < st) red[tid] += red[tid + st]; __syncthreads(); }
  if (tid == 0) {
    d_out[(long)b * SEQ + t] = (float)idx;             // tours (as float)
    d_out[(long)BSZ * SEQ + b] += -__logf(red[0]);     // sum of logps
    mask[b * TOKN + idx] = 1;
  }
  h_t[(long)b * DMODEL + tid]      = h_enc[((long)b * TOKN + idx) * DMODEL + tid]      + pe[(t + 1) * DMODEL + tid];
  h_t[(long)b * DMODEL + 64 + tid] = h_enc[((long)b * TOKN + idx) * DMODEL + 64 + tid] + pe[(t + 1) * DMODEL + 64 + tid];
}

// ---------------------------------------------------------------------------
// PE table (rows 0..50); decoder init (h_t0, mask0, zero logp region)
// ---------------------------------------------------------------------------
__global__ void init_pe_kernel(float* __restrict__ pe)
{
  int i = blockIdx.x * blockDim.x + threadIdx.x;
  if (i < TOKN * DMODEL) {
    int p = i >> 7, d = i & 127;
    int pair = d >> 1;
    float div = __expf((float)(2 * pair) * (-9.210340371976184f / (float)DMODEL));
    float a = (float)p * div;
    pe[i] = (d & 1) ? __cosf(a) : __sinf(a);
  }
}

__global__ void init_dec_kernel(const float* __restrict__ h_enc,
                                const float* __restrict__ pe,
                                float* __restrict__ h_t,
                                int* __restrict__ mask,
                                float* __restrict__ d_out)
{
  int i = blockIdx.x * blockDim.x + threadIdx.x;
  if (i < BSZ * DMODEL) {
    int b = i >> 7, d = i & 127;
    h_t[i] = h_enc[((long)b * TOKN + SEQ) * DMODEL + d] + pe[d];
  }
  if (i < BSZ * TOKN) mask[i] = ((i % TOKN) == SEQ) ? 1 : 0;
  if (i < BSZ) d_out[(long)BSZ * SEQ + i] = 0.f;
}

// ---------------------------------------------------------------------------
// Host orchestration
// ---------------------------------------------------------------------------
extern "C" void kernel_launch(void* const* d_in, const int* in_sizes, int n_in,
                              void* d_out, int out_size, void* d_ws, size_t ws_size,
                              hipStream_t stream)
{
  (void)in_sizes; (void)n_in; (void)out_size; (void)ws_size;
  auto F = [&](int i) { return (const float*)d_in[i]; };

  // Input order: top-level dict insertion order (x, params); params flattened
  // as a JAX pytree (dict keys sorted alphabetically; lin dicts give b then w).
  const float* x = F(0);
  int i = 1;
  const float* WK_b = F(i++);       const float* WK_w = F(i++);
  const float* WV_b = F(i++);       const float* WV_w = F(i++);
  const float* Wqf_b = F(i++);      const float* Wqf_w = F(i++);
  // dec[0]
  const float* W0att_b = F(i++);    const float* W0att_w = F(i++);
  const float* W0sa_b  = F(i++);    const float* W0sa_w  = F(i++);
  const float* W1_b    = F(i++);    const float* W1_w    = F(i++);
  const float* W2_b    = F(i++);    const float* W2_w    = F(i++);
  const float* Wksa_b  = F(i++);    const float* Wksa_w  = F(i++);
  const float* Wqatt_b = F(i++);    const float* Wqatt_w = F(i++);
  const float* Wqsa_b  = F(i++);    const float* Wqsa_w  = F(i++);
  const float* Wvsa_b  = F(i++);    const float* Wvsa_w  = F(i++);
  const float* ln_att_b = F(i++);   const float* ln_att_g = F(i++);
  const float* ln_mlp_b = F(i++);   const float* ln_mlp_g = F(i++);
  const float* ln_sa_b  = F(i++);   const float* ln_sa_g  = F(i++);
  // enc[6] : in_proj(b,w), lin1(b,w), lin2(b,w), ln1_b, ln1_g, ln2_b, ln2_g, out(b,w)
  const int enc_base = i;
  i += 6 * 12;
  const float* emb_b = F(i++);      const float* emb_w = F(i++);
  const float* start = F(i++);

  // Workspace layout (floats)
  float* ws = (float*)d_ws;
  size_t o = 0;
  float* h    = ws + o; o += (size_t)TOK * DMODEL;   // encoder hidden / h_enc
  float* tmpA = ws + o; o += (size_t)TOK * FF;       // qkv (384) / ff1 (512)
  float* tmpB = ws + o; o += (size_t)TOK * DMODEL;
  float* tmpC = ws + o; o += (size_t)TOK * DMODEL;
  float* Katt = ws + o; o += (size_t)TOK * 256;
  float* Vatt = ws + o; o += (size_t)TOK * 256;
  float* Kc   = ws + o; o += (size_t)BSZ * SEQ * DMODEL;
  float* Vc   = ws + o; o += (size_t)BSZ * SEQ * DMODEL;
  float* h_t  = ws + o; o += (size_t)BSZ * DMODEL;
  float* qb   = ws + o; o += (size_t)BSZ * DMODEL;
  float* ob   = ws + o; o += (size_t)BSZ * DMODEL;
  float* pb   = ws + o; o += (size_t)BSZ * DMODEL;
  float* ffb  = ws + o; o += (size_t)BSZ * DMODEL;
  float* pe   = ws + o; o += (size_t)TOKN * DMODEL;
  int*   mask = (int*)(ws + o);

  float* out = (float*)d_out;

  auto gemm = [&](const float* A, int lda, const float* W, int ldw, const float* bias,
                  float* C, int ldc, int M, int N, int K, int act) {
    dim3 g(M / 64, N / 64);
    gemm_bf16_wmma<<<g, 256, 0, stream>>>(A, lda, W, ldw, bias, C, ldc, K, act);
  };

  // ---- constants + embedding ----
  init_pe_kernel<<<(TOKN * DMODEL + 255) / 256, 256, 0, stream>>>(pe);
  embed_kernel<<<((long)TOK * DMODEL + 255) / 256, 256, 0, stream>>>(x, emb_w, emb_b, start, h);

  // ---- encoder: 6 layers ----
  for (int l = 0; l < 6; ++l) {
    const int eb = enc_base + l * 12;
    const float* ip_b = F(eb + 0);  const float* ip_w = F(eb + 1);
    const float* l1_b = F(eb + 2);  const float* l1_w = F(eb + 3);
    const float* l2_b = F(eb + 4);  const float* l2_w = F(eb + 5);
    const float* g1b = F(eb + 6);   const float* g1g = F(eb + 7);
    const float* g2b = F(eb + 8);   const float* g2g = F(eb + 9);
    const float* ou_b = F(eb + 10); const float* ou_w = F(eb + 11);

    gemm(h, DMODEL, ip_w, 3 * DMODEL, ip_b, tmpA, 3 * DMODEL, TOK, 3 * DMODEL, DMODEL, 0);
    attn_enc_kernel<<<BSZ * NHEAD, 64, 0, stream>>>(tmpA, tmpB);
    gemm(tmpB, DMODEL, ou_w, DMODEL, ou_b, tmpC, DMODEL, TOK, DMODEL, DMODEL, 0);
    add_ln_kernel<<<TOK, 128, 0, stream>>>(h, tmpC, g1g, g1b);
    gemm(h, DMODEL, l1_w, FF, l1_b, tmpA, FF, TOK, FF, DMODEL, 1 /*relu*/);
    gemm(tmpA, FF, l2_w, DMODEL, l2_b, tmpB, DMODEL, TOK, DMODEL, FF, 0);
    add_ln_kernel<<<TOK, 128, 0, stream>>>(h, tmpB, g2g, g2b);
  }

  // ---- cross-attention memory ----
  gemm(h, DMODEL, WK_w, 256, WK_b, Katt, 256, TOK, 256, DMODEL, 0);
  gemm(h, DMODEL, WV_w, 256, WV_b, Vatt, 256, TOK, 256, DMODEL, 0);

  // ---- decoder init ----
  init_dec_kernel<<<(BSZ * DMODEL + 255) / 256, 256, 0, stream>>>(h, pe, h_t, mask, out);

  // ---- autoregressive decode: 50 steps ----
  for (int t = 0; t < SEQ; ++t) {
    gemm(h_t, DMODEL, Wqsa_w, DMODEL, Wqsa_b, qb, DMODEL, BSZ, DMODEL, DMODEL, 0);
    gemm(h_t, DMODEL, Wksa_w, DMODEL, Wksa_b, Kc + (long)t * DMODEL, SEQ * DMODEL, BSZ, DMODEL, DMODEL, 0);
    gemm(h_t, DMODEL, Wvsa_w, DMODEL, Wvsa_b, Vc + (long)t * DMODEL, SEQ * DMODEL, BSZ, DMODEL, DMODEL, 0);
    dec_self_attn<<<BSZ * NHEAD, 64, 0, stream>>>(qb, Kc, Vc, ob, t);
    gemm(ob, DMODEL, W0sa_w, DMODEL, W0sa_b, pb, DMODEL, BSZ, DMODEL, DMODEL, 0);
    add_ln_kernel<<<BSZ, 128, 0, stream>>>(h_t, pb, ln_sa_g, ln_sa_b);

    gemm(h_t, DMODEL, Wqatt_w, DMODEL, Wqatt_b, qb, DMODEL, BSZ, DMODEL, DMODEL, 0);
    dec_cross_attn<<<BSZ * NHEAD, 64, 0, stream>>>(qb, Katt, Vatt, mask, ob);
    gemm(ob, DMODEL, W0att_w, DMODEL, W0att_b, pb, DMODEL, BSZ, DMODEL, DMODEL, 0);
    add_ln_kernel<<<BSZ, 128, 0, stream>>>(h_t, pb, ln_att_g, ln_att_b);

    gemm(h_t, DMODEL, W1_w, DMODEL, W1_b, ffb, DMODEL, BSZ, DMODEL, DMODEL, 1 /*relu*/);
    gemm(ffb, DMODEL, W2_w, DMODEL, W2_b, pb, DMODEL, BSZ, DMODEL, DMODEL, 0);
    add_ln_kernel<<<BSZ, 128, 0, stream>>>(h_t, pb, ln_mlp_g, ln_mlp_b);

    gemm(h_t, DMODEL, Wqf_w, DMODEL, Wqf_b, qb, DMODEL, BSZ, DMODEL, DMODEL, 0);
    pointer_kernel<<<BSZ, 64, 0, stream>>>(qb, Katt, mask, h, pe, h_t, out, t);
  }
}
